// MultiHead_27642409517465
// MI455X (gfx1250) — compile-verified, matrix operand order checked by reading, not run on hardware
//
#include <hip/hip_runtime.h>
#include <hip/hip_bf16.h>

// ---------------------------------------------------------------------------
// Multi-head causal attention (B=2, T=2048, C=1024, H=16, d=64) for gfx1250.
// Compute-bound (~95 GFLOP vs ~55MB traffic) -> bf16 WMMA everywhere,
// f32 accumulation, flash-style fused attention (no T*T logits in HBM).
// Data movement: GLOBAL_LOAD_ASYNC_TO_LDS_B128 (ASYNCcnt) + global_prefetch.
// ---------------------------------------------------------------------------

#define BB   2
#define TT   2048
#define CC   1024
#define HH   16
#define DD   64
#define N3C  (3 * CC)          // 3072
#define MM   (BB * TT)         // 4096

typedef __attribute__((ext_vector_type(16))) __bf16 v16bf;
typedef __attribute__((ext_vector_type(8)))  float  v8f;
typedef int v4i_ __attribute__((vector_size(16)));

#if defined(__has_builtin)
#if __has_builtin(__builtin_amdgcn_global_load_async_to_lds_b128) && \
    __has_builtin(__builtin_amdgcn_s_wait_asynccnt)
#define USE_ASYNC 1
#endif
#endif
#ifndef USE_ASYNC
#define USE_ASYNC 0
#endif

struct Frag16 {
    union { v16bf v; unsigned int u[8]; };
};

__device__ __forceinline__ unsigned short f2bf(float f) {
    unsigned int u = __float_as_uint(f);
    unsigned int r = u + 0x7FFFu + ((u >> 16) & 1u);   // round-to-nearest-even
    return (unsigned short)(r >> 16);
}

#if USE_ASYNC
// One 16-byte lane-wise async copy global -> LDS (tracked by ASYNCcnt).
__device__ __forceinline__ void async_cp16(const void* g, void* l) {
    __builtin_amdgcn_global_load_async_to_lds_b128(
        (__attribute__((address_space(1))) v4i_*)g,
        (__attribute__((address_space(3))) v4i_*)l, 0, 0);
}
#endif

// Load a 16x32 bf16 WMMA fragment (A layout; also used for B when the tile is
// stored [n][k] so that element (k,n) = base[n*stride + k]).
// Layout per ISA 7.12.2: lane L holds row (idx + L&15); half = L>>4 selects
// K groups {0..7 | 8..15}; VGPR v holds K pair  (v>>2)*16 + half*8 + (v&3)*2.
__device__ __forceinline__ Frag16 load_frag(const unsigned short* base,
                                            int stride, int idx, int k0) {
    Frag16 f;
    const int lane = threadIdx.x & 31;
    const unsigned short* p =
        base + (idx + (lane & 15)) * stride + k0 + ((lane >> 4) << 3);
#pragma unroll
    for (int v = 0; v < 8; ++v) {
        f.u[v] = *(const unsigned int*)(p + ((v >> 2) << 4) + ((v & 3) << 1));
    }
    return f;
}

// ---------------------------------------------------------------------------
// f32 -> bf16 downconvert
// ---------------------------------------------------------------------------
__global__ __launch_bounds__(256) void cvt_f32_bf16(const float* __restrict__ src,
                                                    unsigned short* __restrict__ dst,
                                                    int n) {
    int i = blockIdx.x * 256 + threadIdx.x;
    if (i < n) dst[i] = f2bf(src[i]);
}

// ---------------------------------------------------------------------------
// QKV projection:  qkv[m,o] = sum_k x[m,k] * W[o,k]
// Block tile 128(M) x 128(N); 8 waves: wave w owns rows [w*16, w*16+16),
// 8 accumulators covering N columns 0..127.  K staged 32 wide through LDS.
// Epilogue scatters into Q/K/V (B,H,T,d) bf16; Q pre-scaled by d^-0.5.
// ---------------------------------------------------------------------------
__global__ __launch_bounds__(256) void qkv_gemm(const unsigned short* __restrict__ X,
                                                const unsigned short* __restrict__ W,
                                                unsigned short* __restrict__ Qo,
                                                unsigned short* __restrict__ Ko,
                                                unsigned short* __restrict__ Vo) {
    __shared__ __align__(16) unsigned short As[128][32];  // x tile   (8 KB)
    __shared__ __align__(16) unsigned short Bs[128][32];  // W rows   (8 KB)

    const int tid   = threadIdx.x;
    const int wave  = tid >> 5;
    const int lane  = tid & 31;
    const int half  = lane >> 4;
    const int nlo   = lane & 15;
    const int Mbase = blockIdx.y * 128;
    const int Nbase = blockIdx.x * 128;

    v8f acc[8];
#pragma unroll
    for (int j = 0; j < 8; ++j) acc[j] = {};

    const uint4* __restrict__ X4 = (const uint4*)X;   // 8 bf16 per uint4
    const uint4* __restrict__ W4 = (const uint4*)W;
    uint4* As4 = (uint4*)&As[0][0];
    uint4* Bs4 = (uint4*)&Bs[0][0];

    // Per-thread source indices for the two A chunks and two B chunks.
    const int rowA0 = tid >> 2,            cA0 = tid & 3;
    const int rowA1 = (tid + 256) >> 2,    cA1 = tid & 3;   // +256 keeps c same
    for (int k0 = 0; k0 < CC; k0 += 32) {
        __syncthreads();
        const int kq = k0 >> 3;
#if USE_ASYNC
        async_cp16(&X4[(Mbase + rowA0) * (CC / 8) + kq + cA0], &As4[tid]);
        async_cp16(&X4[(Mbase + rowA1) * (CC / 8) + kq + cA1], &As4[tid + 256]);
        async_cp16(&W4[(Nbase + rowA0) * (CC / 8) + kq + cA0], &Bs4[tid]);
        async_cp16(&W4[(Nbase + rowA1) * (CC / 8) + kq + cA1], &Bs4[tid + 256]);
        if (k0 + 32 < CC) {   // prefetch next K-slab
            __builtin_prefetch(&X4[(Mbase + rowA0) * (CC / 8) + kq + 4 + cA0], 0, 1);
            __builtin_prefetch(&W4[(Nbase + rowA0) * (CC / 8) + kq + 4 + cA0], 0, 1);
        }
        __builtin_amdgcn_s_wait_asynccnt(0);
#else
        As4[tid]       = X4[(Mbase + rowA0) * (CC / 8) + kq + cA0];
        As4[tid + 256] = X4[(Mbase + rowA1) * (CC / 8) + kq + cA1];
        Bs4[tid]       = W4[(Nbase + rowA0) * (CC / 8) + kq + cA0];
        Bs4[tid + 256] = W4[(Nbase + rowA1) * (CC / 8) + kq + cA1];
#endif
        __syncthreads();

        Frag16 a = load_frag(&As[0][0], 32, wave * 16, 0);
#pragma unroll
        for (int j = 0; j < 8; ++j) {
            Frag16 b = load_frag(&Bs[0][0], 32, j * 16, 0);
            acc[j] = __builtin_amdgcn_wmma_f32_16x16x32_bf16(
                false, a.v, false, b.v, (short)0, acc[j], false, false);
        }
    }

    // Epilogue: scatter into per-head Q/K/V, scale Q by 1/sqrt(d)=0.125.
#pragma unroll
    for (int j = 0; j < 8; ++j) {
#pragma unroll
        for (int r = 0; r < 8; ++r) {
            int m = Mbase + wave * 16 + r + half * 8;
            int o = Nbase + j * 16 + nlo;
            int b = m >> 11;            // / T
            int t = m & (TT - 1);
            int h = o / 192;
            int rem = o - h * 192;
            int part = rem >> 6;        // 0:Q 1:K 2:V
            int col = rem & 63;
            float val = acc[j][r];
            if (part == 0) val *= 0.125f;
            unsigned short* dst = (part == 0) ? Qo : (part == 1) ? Ko : Vo;
            dst[(((b * HH + h) * TT + t) << 6) + col] = f2bf(val);
        }
    }
}

// ---------------------------------------------------------------------------
// Flash attention: block = (b,h, 128 query rows); 8 waves x 16 rows.
// Iterate 64-key tiles up to the causal frontier; online softmax.
// ---------------------------------------------------------------------------
__global__ __launch_bounds__(256) void attn(const unsigned short* __restrict__ Q,
                                            const unsigned short* __restrict__ K,
                                            const unsigned short* __restrict__ V,
                                            float* __restrict__ out) {
    __shared__ __align__(16) unsigned short Qs[128][DD];      // 16 KB
    __shared__ __align__(16) unsigned short Ks[64][DD];       //  8 KB  [key][d]
    __shared__ __align__(16) unsigned short Vt[DD][64];       //  8 KB  [d][key]
    __shared__ __align__(16) unsigned short Ps[8][16][64];    // 16 KB  per-wave P

    const int tid  = threadIdx.x;
    const int wave = tid >> 5;
    const int lane = tid & 31;
    const int half = lane >> 4;
    const int nlo  = lane & 15;

    const int qbase = blockIdx.x * 128;
    const int bh    = blockIdx.y;
    const int b     = bh / HH;
    const int h     = bh - b * HH;

    const unsigned short* qptr    = Q + ((bh * TT + qbase) << 6);
    const unsigned short* kbase_p = K + ((bh * TT) << 6);
    const unsigned short* vbase_p = V + ((bh * TT) << 6);

    // Load Q tile once: 128*64 bf16 = 1024 uint4, 4 per thread.
    {
        const uint4* s = (const uint4*)qptr;
        uint4* d = (uint4*)&Qs[0][0];
#pragma unroll
        for (int q = 0; q < 4; ++q) {
#if USE_ASYNC
            async_cp16(&s[tid + q * 256], &d[tid + q * 256]);
#else
            d[tid + q * 256] = s[tid + q * 256];
#endif
        }
    }
#if USE_ASYNC
    __builtin_amdgcn_s_wait_asynccnt(0);
#endif
    __syncthreads();

    float m_r[8], l_r[8];
    v8f acc[4];
#pragma unroll
    for (int r = 0; r < 8; ++r) { m_r[r] = -INFINITY; l_r[r] = 0.f; }
#pragma unroll
    for (int j = 0; j < 4; ++j) acc[j] = {};

    const int kt_end = (qbase + 127) >> 6;   // inclusive causal frontier
    for (int kt = 0; kt <= kt_end; ++kt) {
        __syncthreads();   // previous iteration done with Ks/Vt
        // K tile: 64*64 bf16 = 512 uint4, 2 per thread (flat copy, async path)
        {
            const uint4* s = (const uint4*)(kbase_p + ((kt * 64) << 6));
            uint4* d = (uint4*)&Ks[0][0];
#if USE_ASYNC
            async_cp16(&s[tid], &d[tid]);
            async_cp16(&s[tid + 256], &d[tid + 256]);
#else
            d[tid] = s[tid];
            d[tid + 256] = s[tid + 256];
#endif
        }
        // V tile, transposed into Vt[d][key]: 2048 uints, 8 per thread
        {
            const unsigned int* s = (const unsigned int*)(vbase_p + ((kt * 64) << 6));
#pragma unroll
            for (int q = 0; q < 8; ++q) {
                int lin = tid + q * 256;
                int kk = lin >> 5;          // key row
                int du = lin & 31;          // d pair
                unsigned int w = s[kk * 32 + du];
                Vt[du * 2][kk]     = (unsigned short)(w & 0xffffu);
                Vt[du * 2 + 1][kk] = (unsigned short)(w >> 16);
            }
        }
        if (kt < kt_end) {   // prefetch next K/V tiles
            __builtin_prefetch(kbase_p + (((kt + 1) * 64) << 6) + tid * 8, 0, 1);
            __builtin_prefetch(vbase_p + (((kt + 1) * 64) << 6) + tid * 8, 0, 1);
        }
#if USE_ASYNC
        __builtin_amdgcn_s_wait_asynccnt(0);
#endif
        __syncthreads();

        // S = Q_tile @ K_tile^T  (16 x 64 per wave)
        v8f S[4];
#pragma unroll
        for (int j = 0; j < 4; ++j) S[j] = {};
#pragma unroll
        for (int dk = 0; dk < DD; dk += 32) {
            Frag16 a = load_frag(&Qs[0][0], DD, wave * 16, dk);
#pragma unroll
            for (int j = 0; j < 4; ++j) {
                Frag16 bf = load_frag(&Ks[0][0], DD, j * 16, dk);
                S[j] = __builtin_amdgcn_wmma_f32_16x16x32_bf16(
                    false, a.v, false, bf.v, (short)0, S[j], false, false);
            }
        }

        // Causal mask + online softmax (rows live in 16-lane halves).
#pragma unroll
        for (int r = 0; r < 8; ++r) {
            const int rowInWave = r + half * 8;
            const int qg = qbase + wave * 16 + rowInWave;
            float mx = -INFINITY;
#pragma unroll
            for (int j = 0; j < 4; ++j) {
                int kg = kt * 64 + j * 16 + nlo;
                float s = S[j][r];
                if (kg > qg) s = -INFINITY;
                S[j][r] = s;
                mx = fmaxf(mx, s);
            }
#pragma unroll
            for (int off = 8; off >= 1; off >>= 1)
                mx = fmaxf(mx, __shfl_xor(mx, off, 32));
            float mn = fmaxf(m_r[r], mx);
            float alpha = __expf(m_r[r] - mn);
            float rs = 0.f;
#pragma unroll
            for (int j = 0; j < 4; ++j) {
                float p = __expf(S[j][r] - mn);
                S[j][r] = p;
                rs += p;
            }
#pragma unroll
            for (int off = 8; off >= 1; off >>= 1)
                rs += __shfl_xor(rs, off, 32);
            l_r[r] = l_r[r] * alpha + rs;
            m_r[r] = mn;
#pragma unroll
            for (int j = 0; j < 4; ++j) acc[j][r] *= alpha;
            // Stage P (C layout -> LDS row-major for A-fragment reload)
#pragma unroll
            for (int j = 0; j < 4; ++j)
                Ps[wave][rowInWave][j * 16 + nlo] = f2bf(S[j][r]);
        }
        __syncthreads();

        // acc += P @ V_tile   (A = P 16x64 over keys, B = Vt[d][key])
#pragma unroll
        for (int dkk = 0; dkk < 64; dkk += 32) {
            Frag16 a = load_frag(&Ps[wave][0][0], 64, 0, dkk);
#pragma unroll
            for (int j = 0; j < 4; ++j) {
                Frag16 bf = load_frag(&Vt[0][0], 64, j * 16, dkk);
                acc[j] = __builtin_amdgcn_wmma_f32_16x16x32_bf16(
                    false, a.v, false, bf.v, (short)0, acc[j], false, false);
            }
        }
    }

    // out[b, t, h*64 + col] = acc / l
#pragma unroll
    for (int r = 0; r < 8; ++r) {
        const int qg = qbase + wave * 16 + r + half * 8;
        const float inv_l = 1.0f / l_r[r];
        float* op = out + (b * TT + qg) * CC + h * DD;
#pragma unroll
        for (int j = 0; j < 4; ++j)
            op[j * 16 + nlo] = acc[j][r] * inv_l;
    }
}

// ---------------------------------------------------------------------------
extern "C" void kernel_launch(void* const* d_in, const int* in_sizes, int n_in,
                              void* d_out, int out_size, void* d_ws, size_t ws_size,
                              hipStream_t stream) {
    const float* x = (const float*)d_in[0];   // (B,T,C) f32
    const float* w = (const float*)d_in[1];   // (3C,C) f32
    float* out = (float*)d_out;               // (B,T,C) f32

    const size_t nX = (size_t)MM * CC;           // 4,194,304
    const size_t nW = (size_t)N3C * CC;          // 3,145,728
    const size_t nH = (size_t)BB * HH * TT * DD; // == nX

    unsigned short* Xb = (unsigned short*)d_ws;
    unsigned short* Wb = Xb + nX;
    unsigned short* Qp = Wb + nW;
    unsigned short* Kp = Qp + nH;
    unsigned short* Vp = Kp + nH;

    cvt_f32_bf16<<<(int)(nX / 256), 256, 0, stream>>>(x, Xb, (int)nX);
    cvt_f32_bf16<<<(int)(nW / 256), 256, 0, stream>>>(w, Wb, (int)nW);

    qkv_gemm<<<dim3(N3C / 128, MM / 128), 256, 0, stream>>>(Xb, Wb, Qp, Kp, Vp);

    attn<<<dim3(TT / 128, BB * HH), 256, 0, stream>>>(Qp, Kp, Vp, out);
}
